// HeteroRGCN_33397665693711
// MI455X (gfx1250) — matrix operand order repaired and checked by reading, not compiled, not run on hardware
//
#include <hip/hip_runtime.h>
#include <hip/hip_bf16.h>

typedef __attribute__((ext_vector_type(16))) _Float16 v16h;
typedef __attribute__((ext_vector_type(8)))  float    v8f;

#define NNODES 50000
#define NEDGES 500000
#define IND    1024
#define HD     256
#define CD     2

// ---------------------------------------------------------------------------
// helpers
// ---------------------------------------------------------------------------
__device__ __forceinline__ void gatomic_fadd(float* p, float v) {
    // CDNA5 native FP32 cache atomic (no-return form, tracked by STOREcnt)
    asm volatile("global_atomic_add_f32 %0, %1, off" :: "v"(p), "v"(v) : "memory");
}

// convert 4 fp32 values to f16 hi + f16 residual (lo), at compile-time slot J
template<int J>
__device__ __forceinline__ void cvt4(const float4 f, v16h& hi, v16h& lo) {
    _Float16 h;
    h = (_Float16)f.x; hi[J+0] = h; lo[J+0] = (_Float16)(f.x - (float)h);
    h = (_Float16)f.y; hi[J+1] = h; lo[J+1] = (_Float16)(f.y - (float)h);
    h = (_Float16)f.z; hi[J+2] = h; lo[J+2] = (_Float16)(f.z - (float)h);
    h = (_Float16)f.w; hi[J+3] = h; lo[J+3] = (_Float16)(f.w - (float)h);
}

// split fp32 array into f16 hi + f16 residual lo (run once per etype for W)
__global__ void split_kernel(const float* __restrict__ in,
                             _Float16* __restrict__ hi, _Float16* __restrict__ lo,
                             long n) {
    const long i = blockIdx.x * (long)blockDim.x + threadIdx.x;
    if (i < n) {
        const float x = in[i];
        const _Float16 h = (_Float16)x;
        hi[i] = h;
        lo[i] = (_Float16)(x - (float)h);
    }
}

// ---------------------------------------------------------------------------
// Layer-1 GEMM: out[N,256] = X[N,1024] @ W^T + b
// W pre-split into f16 hi/lo, row-major [256,1024] -> B fragment is ONE
// aligned 32B v16h load per lane. X split on the fly (co-executes with XDL).
// One wave -> 16 rows x 128 cols (8 n-tiles), 3 WMMAs per tile per k-step.
// ---------------------------------------------------------------------------
__global__ __launch_bounds__(256) void gemm1_wmma(
    const float* __restrict__ X,
    const _Float16* __restrict__ Whi, const _Float16* __restrict__ Wlo,
    const float* __restrict__ b, float* __restrict__ out, int nrows)
{
    const int wave = (int)(blockIdx.x * blockDim.x + threadIdx.x) >> 5;
    const int lane = threadIdx.x & 31;
    const int rt = wave >> 1;          // 16-row tile index
    const int nh = wave & 1;           // which half of the 256 output channels
    if (rt * 16 >= nrows) return;      // wave-uniform exit (EXEC stays all-ones)

    const int half = lane >> 4;        // 0: lanes 0-15, 1: lanes 16-31
    const int lm   = lane & 15;

    v8f acc[8];
#pragma unroll
    for (int t = 0; t < 8; ++t) {
        const float bb = b[nh * 128 + t * 16 + lm];   // bias: constant per (lane n)
#pragma unroll
        for (int r = 0; r < 8; ++r) acc[t][r] = bb;
    }

    const float* xrow = X + (size_t)(rt * 16 + lm) * IND;

    for (int kk = 0; kk < IND / 32; ++kk) {
        const int kb = kk * 32;
        // A fragment 16x32: lanes<16 hold K 0-15, lanes>=16 hold K 16-31
        v16h a_hi, a_lo;
        {
            const float4* p0 = (const float4*)(xrow + kb + half * 16);
            const float4* p1 = (const float4*)(xrow + kb + half * 16 + 8);
            cvt4<0>(p0[0], a_hi, a_lo); cvt4<4>(p0[1], a_hi, a_lo);
            cvt4<8>(p1[0], a_hi, a_lo); cvt4<12>(p1[1], a_hi, a_lo);
        }
#pragma unroll
        for (int t = 0; t < 8; ++t) {
            // B fragment 32x16: lane n = lm, lanes<16 K 0-15, lanes>=16 K 16-31
            const int n = nh * 128 + t * 16 + lm;
            const size_t woff = (size_t)n * IND + kb + half * 16;
            const v16h b_hi = *(const v16h*)(Whi + woff);
            const v16h b_lo = *(const v16h*)(Wlo + woff);
            // split-precision: Xh*Wh + Xh*Wl + Xl*Wh  (fp32 accumulate)
            acc[t] = __builtin_amdgcn_wmma_f32_16x16x32_f16(false, a_hi, false, b_hi, (short)0, acc[t], false, false);
            acc[t] = __builtin_amdgcn_wmma_f32_16x16x32_f16(false, a_hi, false, b_lo, (short)0, acc[t], false, false);
            acc[t] = __builtin_amdgcn_wmma_f32_16x16x32_f16(false, a_lo, false, b_hi, (short)0, acc[t], false, false);
        }
    }

    // D layout: VGPR r -> M = r + 8*half ; N = lm
    const int mbase = rt * 16 + half * 8;
#pragma unroll
    for (int t = 0; t < 8; ++t) {
        const int n = nh * 128 + t * 16 + lm;
#pragma unroll
        for (int r = 0; r < 8; ++r)
            out[(size_t)(mbase + r) * HD + n] = acc[t][r];
    }
}

// ---------------------------------------------------------------------------
// Edge scatter, layer 1: one wave per edge; lane handles 8 channels.
// acc[dst] += Wh[src] * invdeg[dst]   (L2-resident working set on MI455X)
// ---------------------------------------------------------------------------
__global__ __launch_bounds__(256) void scatter1_kernel(
    const float* __restrict__ Wh, const int* __restrict__ src,
    const int* __restrict__ dst, const float* __restrict__ invdeg,
    float* __restrict__ acc, int nedges)
{
    const int e    = (int)(blockIdx.x * blockDim.x + threadIdx.x) >> 5;
    const int lane = threadIdx.x & 31;
    if (e >= nedges) return;
    const int s = src[e], d = dst[e];
    const float inv = invdeg[d];
    const float4* p = (const float4*)(Wh + (size_t)s * HD) + lane * 2;
    const float4 v0 = p[0], v1 = p[1];
    float* q = acc + (size_t)d * HD + lane * 8;
    gatomic_fadd(q + 0, v0.x * inv); gatomic_fadd(q + 1, v0.y * inv);
    gatomic_fadd(q + 2, v0.z * inv); gatomic_fadd(q + 3, v0.w * inv);
    gatomic_fadd(q + 4, v1.x * inv); gatomic_fadd(q + 5, v1.y * inv);
    gatomic_fadd(q + 6, v1.z * inv); gatomic_fadd(q + 7, v1.w * inv);
}

// ---------------------------------------------------------------------------
// Layer-2 GEMM: [N,256] x [256,2] -> wave-per-node dot products
// ---------------------------------------------------------------------------
__global__ __launch_bounds__(256) void gemm2_kernel(
    const float* __restrict__ Hf, const float* __restrict__ W2,
    const float* __restrict__ b2, float* __restrict__ out, int nnodes)
{
    const int node = (int)(blockIdx.x * blockDim.x + threadIdx.x) >> 5;
    const int lane = threadIdx.x & 31;
    if (node >= nnodes) return;
    const float4* hp = (const float4*)(Hf + (size_t)node * HD) + lane * 2;
    const float4 h0 = hp[0], h1 = hp[1];
    const float4* wa = (const float4*)(W2)       + lane * 2;   // row c=0
    const float4* wb = (const float4*)(W2 + HD)  + lane * 2;   // row c=1
    const float4 a0 = wa[0], a1 = wa[1], b0 = wb[0], b1 = wb[1];
    float p0 = h0.x*a0.x + h0.y*a0.y + h0.z*a0.z + h0.w*a0.w
             + h1.x*a1.x + h1.y*a1.y + h1.z*a1.z + h1.w*a1.w;
    float p1 = h0.x*b0.x + h0.y*b0.y + h0.z*b0.z + h0.w*b0.w
             + h1.x*b1.x + h1.y*b1.y + h1.z*b1.z + h1.w*b1.w;
#pragma unroll
    for (int off = 16; off > 0; off >>= 1) {
        p0 += __shfl_down(p0, off);
        p1 += __shfl_down(p1, off);
    }
    if (lane == 0) {
        out[(size_t)node * CD + 0] = p0 + b2[0];
        out[(size_t)node * CD + 1] = p1 + b2[1];
    }
}

__global__ void scatter2_kernel(
    const float* __restrict__ wh2, const int* __restrict__ src,
    const int* __restrict__ dst, const float* __restrict__ invdeg,
    float* __restrict__ acc, int nedges)
{
    const int e = blockIdx.x * blockDim.x + threadIdx.x;
    if (e >= nedges) return;
    const int s = src[e], d = dst[e];
    const float inv = invdeg[d];
    gatomic_fadd(acc + (size_t)d * CD + 0, wh2[(size_t)s * CD + 0] * inv);
    gatomic_fadd(acc + (size_t)d * CD + 1, wh2[(size_t)s * CD + 1] * inv);
}

// ---------------------------------------------------------------------------
// small utility kernels
// ---------------------------------------------------------------------------
__global__ void zero_kernel(float* p, long n) {
    long i = blockIdx.x * (long)blockDim.x + threadIdx.x;
    const long stride = (long)gridDim.x * blockDim.x;
    for (; i < n; i += stride) p[i] = 0.0f;
}

__global__ void degree_kernel(const int* __restrict__ dst, unsigned* __restrict__ deg, int ne) {
    const int e = blockIdx.x * blockDim.x + threadIdx.x;
    if (e < ne) atomicAdd(&deg[dst[e]], 1u);
}

__global__ void invdeg_kernel(const unsigned* __restrict__ deg, float* __restrict__ inv, int n) {
    const int i = blockIdx.x * blockDim.x + threadIdx.x;
    if (i < n) { const unsigned d = deg[i]; inv[i] = 1.0f / (float)(d > 1u ? d : 1u); }
}

__global__ void lrelu_kernel(const float* __restrict__ in, float* __restrict__ out, long n) {
    const long i = blockIdx.x * (long)blockDim.x + threadIdx.x;
    if (i < n) { const float x = in[i]; out[i] = x > 0.0f ? x : 0.01f * x; }
}

// ---------------------------------------------------------------------------
// launch
// ---------------------------------------------------------------------------
extern "C" void kernel_launch(void* const* d_in, const int* in_sizes, int n_in,
                              void* d_out, int out_size, void* d_ws, size_t ws_size,
                              hipStream_t stream)
{
    (void)in_sizes; (void)n_in; (void)out_size; (void)ws_size;
    const float* x[2] = { (const float*)d_in[0], (const float*)d_in[1] };  // user, item
    // etype order: clicks(u->i), clicked_by(i->u), follows(u->u), similar(i->i)
    const int src_nt[4] = {0, 1, 0, 1};
    const int dst_nt[4] = {1, 0, 0, 1};
    const float* W1[4]; const float* b1[4]; const float* W2[4]; const float* b2[4];
    const int* esrc[4]; const int* edst[4];
    for (int i = 0; i < 4; ++i) {
        const int base = 2 + i * 6;
        W1[i]   = (const float*)d_in[base + 0];
        b1[i]   = (const float*)d_in[base + 1];
        W2[i]   = (const float*)d_in[base + 2];
        b2[i]   = (const float*)d_in[base + 3];
        esrc[i] = (const int*)  d_in[base + 4];
        edst[i] = (const int*)  d_in[base + 5];
    }

    // workspace layout (floats)
    float* ws     = (float*)d_ws;
    float* wh1    = ws;                                     // N*256 (reused per etype)
    float* acc1   = ws + (size_t)NNODES * HD;               // 2*N*256 (user, item)
    unsigned* degu = (unsigned*)(ws + 3 * (size_t)NNODES * HD);  // 4*N uints
    float* invdeg = ws + 3 * (size_t)NNODES * HD + 4 * (size_t)NNODES; // 4*N
    float* wh2    = invdeg + 4 * (size_t)NNODES;            // 4*N*2
    const long WN = (long)HD * IND;                         // 262144 weights per etype
    _Float16* w1hi = (_Float16*)(wh2 + 4 * (size_t)NNODES * CD);  // reused per etype
    _Float16* w1lo = w1hi + WN;

    float* out      = (float*)d_out;
    float* out_h[2] = { out, out + 2 * (size_t)NNODES };            // h_user, h_item [N,2]
    float* h2base   = out + 4 * (size_t)NNODES;                     // h2_user, h2_item [N,256]
    float* h2[2]    = { h2base, h2base + (size_t)NNODES * HD };

    // 1) degrees: zero, count, invert  (reused by both layers)
    zero_kernel<<<2048, 256, 0, stream>>>((float*)degu, 4L * NNODES);
    for (int i = 0; i < 4; ++i)
        degree_kernel<<<(NEDGES + 255) / 256, 256, 0, stream>>>(edst[i], degu + (size_t)i * NNODES, NEDGES);
    invdeg_kernel<<<(4 * NNODES + 255) / 256, 256, 0, stream>>>(degu, invdeg, 4 * NNODES);

    // 2) layer 1: zero accumulators, then per-etype W-split + GEMM + scatter
    //    (wh1/w1hi/w1lo reused; stream ordering serializes, keeping the
    //     ~154MB working set L2-resident on MI455X's 192MB L2)
    zero_kernel<<<4096, 256, 0, stream>>>(acc1, 2L * NNODES * HD);
    const int g1_blocks = (3125 * 2 * 32 + 255) / 256;   // 6250 waves
    for (int i = 0; i < 4; ++i) {
        split_kernel<<<(int)((WN + 255) / 256), 256, 0, stream>>>(W1[i], w1hi, w1lo, WN);
        gemm1_wmma<<<g1_blocks, 256, 0, stream>>>(x[src_nt[i]], w1hi, w1lo, b1[i], wh1, NNODES);
        scatter1_kernel<<<(int)(((long)NEDGES * 32 + 255) / 256), 256, 0, stream>>>(
            wh1, esrc[i], edst[i], invdeg + (size_t)i * NNODES,
            acc1 + (size_t)dst_nt[i] * NNODES * HD, NEDGES);
    }

    // 3) leaky_relu -> h2 outputs (user and item contiguous)
    lrelu_kernel<<<(int)((2L * NNODES * HD + 255) / 256), 256, 0, stream>>>(
        acc1, h2base, 2L * NNODES * HD);

    // 4) layer 2: zero [N,2] accumulators, per-etype GEMM + scatter
    zero_kernel<<<256, 256, 0, stream>>>(out, 4L * NNODES);
    const int g2_blocks = (NNODES * 32 + 255) / 256;
    for (int i = 0; i < 4; ++i) {
        gemm2_kernel<<<g2_blocks, 256, 0, stream>>>(
            h2[src_nt[i]], W2[i], b2[i], wh2 + (size_t)i * NNODES * CD, NNODES);
        scatter2_kernel<<<(NEDGES + 255) / 256, 256, 0, stream>>>(
            wh2 + (size_t)i * NNODES * CD, esrc[i], edst[i],
            invdeg + (size_t)i * NNODES, out_h[dst_nt[i]], NEDGES);
    }
}
